// ConditionalRandomField_81209241632769
// MI455X (gfx1250) — compile-verified, matrix operand order checked by reading, not run on hardware
//
#include <hip/hip_runtime.h>
#include <hip/hip_bf16.h>
#include <stdint.h>

// Problem constants (from reference): B=256, S=2048, T=64
#define B_    256
#define S_    2048
#define T_    64
#define DEPTH 4   // async-prefetch lookahead (emission tiles in flight)
#define NBUF  5   // ring slots = DEPTH+1 so refill never collides with reader

#define LOG2E 1.4426950408889634f
#define LN2   0.6931471805599453f

typedef __attribute__((ext_vector_type(16))) _Float16 v16h;
typedef __attribute__((ext_vector_type(8)))  _Float16 v8h;
typedef __attribute__((ext_vector_type(8)))  float    v8f;

// K-index mapping for 16-bit 16x32 WMMA A/B fragments (CDNA5 ISA §7.12.2):
// VGPR v holds halves 2v,2v+1; lanes16-31 offset K by 8; VGPR4-7 offset K by 16.
__device__ __forceinline__ int kmap(int h, int hi) {
  int v = h >> 1, lo = h & 1;
  return 16 * (v >> 2) + 2 * (v & 3) + lo + 8 * hi;
}

// Fast hardware transcendentals (operands guaranteed normal-range here).
__device__ __forceinline__ float fexp(float x) { return __builtin_amdgcn_exp2f(x * LOG2E); }
__device__ __forceinline__ float flog(float x) { return __builtin_amdgcn_logf(x) * LN2; }

// CDNA5 async global->LDS copy (16B per lane), tracked with ASYNCcnt.
__device__ __forceinline__ void async_tile_load(const float* g, unsigned lds_off) {
  asm volatile("global_load_async_to_lds_b128 %0, %1, off"
               :: "v"(lds_off), "v"(g) : "memory");
}
#define WAIT_ASYNC(n) asm volatile("s_wait_asynccnt " #n ::: "memory")

// ---------------------------------------------------------------------------
// Denominator: CRF forward scan. 16 batch rows per block, 4 waves, each wave
// owns one 16-wide N-tile; K=64 contraction = 2 chained f16 WMMAs per step.
// alpha lives in registers in WMMA D-layout (v8f per lane); LDS is only used
// for the f16 A-fragment staging, the rowmax cross-lane max, and emissions.
// ---------------------------------------------------------------------------
__global__ __launch_bounds__(128)
void crf_forward_wmma(const float* __restrict__ inputs,
                      const unsigned char* __restrict__ mask,
                      const float* __restrict__ trans,
                      const float* __restrict__ start_t,
                      const float* __restrict__ end_t,
                      float* __restrict__ den_out)
{
  __shared__ __align__(64) float rowmaxbuf[2][16];   // fed by ds_max_num_f32
  __shared__ __align__(64) float maskrow[16];
  __shared__ _Float16 pbuf[16 * 72];                 // stride 72 halves (144B), conflict-free
  __shared__ float    ebuf[NBUF][16 * 68];           // emission ring, row stride 68

  const int tid   = threadIdx.x;
  const int lane  = tid & 31;
  const int wv    = tid >> 5;                // wave id == N-tile id (0..3)
  const int b0    = blockIdx.x * 16;         // first batch row of this block
  const int hi    = lane >> 4;               // lane group (0: lanes0-15, 1: lanes16-31)
  const int mrow  = lane & 15;               // A-fragment M row
  const int ncol  = 16 * wv + (lane & 15);   // output/B column this lane owns
  const int mbase = 8 * hi;                  // first of the 8 alpha rows this lane owns

  // ---- reset rowmax accumulators ----
  if (tid < 16) { rowmaxbuf[0][tid] = -3.0e38f; rowmaxbuf[1][tid] = -3.0e38f; }
  __syncthreads();

  // ---- alpha0 = start + logits[:,0,:] straight into registers (D-layout) ----
  v8f areg;
  {
    float st = start_t[ncol];
    #pragma unroll
    for (int r = 0; r < 8; ++r) {
      areg[r] = st + inputs[((size_t)(b0 + mbase + r) * S_) * T_ + ncol];
      atomicMax(&rowmaxbuf[1][mbase + r], areg[r]);   // step s=1 uses buffer 1
    }
  }

  // ---- B fragments: exp(transitions) columns, held in VGPRs for whole scan ----
  v16h bf0, bf1;
  #pragma unroll
  for (int h = 0; h < 16; ++h) {
    int K = kmap(h, hi);
    bf0[h] = (_Float16)fexp(trans[(K)      * T_ + ncol]);   // K = 0..31
    bf1[h] = (_Float16)fexp(trans[(K + 32) * T_ + ncol]);   // K = 32..63
  }

  // ---- async prefetch of emission tiles: 16x64 tile, 2 x b128 per thread ----
  auto issue = [&](int s) {
    int buf = (s - 1) % NBUF;
    #pragma unroll
    for (int c2 = 0; c2 < 2; ++c2) {
      int f   = (tid + 128 * c2) * 4;                // float index in 16x64 tile
      int r   = f >> 6, col = f & 63;
      const float* g = inputs + ((size_t)(b0 + r) * S_ + s) * T_ + col;
      unsigned loff  = (unsigned)(uintptr_t)&ebuf[buf][r * 68 + col];
      async_tile_load(g, loff);
    }
  };
  for (int s = 1; s <= DEPTH && s < S_; ++s) issue(s);
  __syncthreads();

  // ---- sequential scan over steps 1..S-1 (2 barriers per step) ----
  for (int s = 1; s < S_; ++s) {
    // drain own ASYNCcnt to steady-state depth; barrier publishes async data,
    // previous rowmax atomics, and retires all readers of pbuf/ebuf slots.
    if (s + DEPTH <= S_ - 1) { WAIT_ASYNC(6); } else { WAIT_ASYNC(0); }
    __syncthreads();

    // phase A: hoist emission column into registers (latency overlaps the exps;
    // completion is covered by the pre-barrier dscnt wait, so phase B has no
    // LDS stalls inside the log/blend chain).
    v8f emv;
    {
      const float* em = &ebuf[(s - 1) % NBUF][0];
      #pragma unroll
      for (int r = 0; r < 8; ++r) emv[r] = em[(mbase + r) * 68 + ncol];
    }
    if (tid < 16) {
      float mk = mask[(size_t)(b0 + tid) * S_ + s] ? 1.0f : 0.0f;
      maskrow[tid] = mk;
      // masked-off rows keep alpha, hence keep their rowmax (branchless blend)
      float cur = rowmaxbuf[s & 1][tid];
      rowmaxbuf[(s + 1) & 1][tid] = fmaf(mk, -3.0e38f - cur, cur);
    }
    v8f rm8 = *(const v8f*)&rowmaxbuf[s & 1][mbase];
    #pragma unroll
    for (int r = 0; r < 8; ++r)
      pbuf[(mbase + r) * 72 + ncol] = (_Float16)fexp(areg[r] - rm8[r]);
    __syncthreads();

    // phase B: S = P x exp(T) via two chained WMMAs per wave
    const _Float16* prow = &pbuf[mrow * 72 + 8 * hi];
    v8h p0 = *(const v8h*)(prow);                      // K 0..7   (+8 if hi)
    v8h p1 = *(const v8h*)(prow + 16);                 // K 16..23 (+8 if hi)
    v8h p2 = *(const v8h*)(prow + 32);                 // K 32..39 (+8 if hi)
    v8h p3 = *(const v8h*)(prow + 48);                 // K 48..55 (+8 if hi)
    v16h a0 = __builtin_shufflevector(p0, p1, 0,1,2,3,4,5,6,7,8,9,10,11,12,13,14,15);
    v16h a1 = __builtin_shufflevector(p2, p3, 0,1,2,3,4,5,6,7,8,9,10,11,12,13,14,15);
    v8f c = {};
    c = __builtin_amdgcn_wmma_f32_16x16x32_f16(false, a0, false, bf0, (short)0, c, false, false);
    c = __builtin_amdgcn_wmma_f32_16x16x32_f16(false, a1, false, bf1, (short)0, c, false, false);

    // branchless un-rescale + emission + arithmetic mask-blend + fused rowmax.
    // blend as fma(mk, nv-old, old) so the compiler cannot re-introduce
    // per-element EXEC branches (mk is exactly 0.0 or 1.0).
    v8f mk8 = *(const v8f*)&maskrow[mbase];
    float* rbnext = &rowmaxbuf[(s + 1) & 1][0];
    #pragma unroll
    for (int r = 0; r < 8; ++r) {
      float nv  = rm8[r] + flog(c[r]) + emv[r];
      float val = fmaf(mk8[r], nv - areg[r], areg[r]);
      areg[r] = val;
      atomicMax(&rbnext[mbase + r], val);   // masked rows: val <= preset rowmax, no-op
    }
    // refill the slot read last step (never the slot read this step)
    int sn = s + DEPTH;
    if (sn <= S_ - 1) issue(sn);
  }

  // ---- stage alpha to LDS once, then log_den[b] = LSE(alpha + end_t) ----
  __syncthreads();
  float* afin = &ebuf[0][0];                 // reuse ring slot 0 (16x68 floats)
  #pragma unroll
  for (int r = 0; r < 8; ++r) afin[(mbase + r) * 68 + ncol] = areg[r];
  __syncthreads();
  if (tid < 16) {
    float m = -3.0e38f;
    for (int i = 0; i < 64; ++i) m = fmaxf(m, afin[tid * 68 + i] + end_t[i]);
    float sum = 0.0f;
    for (int i = 0; i < 64; ++i) sum += fexp(afin[tid * 68 + i] + end_t[i] - m);
    den_out[b0 + tid] = m + flog(sum);
  }
}

// ---------------------------------------------------------------------------
// Numerator: joint likelihood of the gold path. One block per batch row.
// ---------------------------------------------------------------------------
__global__ __launch_bounds__(256)
void crf_joint(const float* __restrict__ inputs,
               const int* __restrict__ tags,
               const unsigned char* __restrict__ mask,
               const float* __restrict__ trans,
               const float* __restrict__ start_t,
               const float* __restrict__ end_t,
               float* __restrict__ num_out)
{
  __shared__ float sred[256];
  __shared__ int   cred[256];
  const int b = blockIdx.x, tid = threadIdx.x;
  const int*           tg = tags + (size_t)b * S_;
  const unsigned char* mk = mask + (size_t)b * S_;
  const float*         lg = inputs + (size_t)b * S_ * T_;

  float acc = 0.0f; int cnt = 0;
  for (int s = tid; s < S_; s += 256) {
    int   t  = tg[s];
    float mf = mk[s] ? 1.0f : 0.0f;
    cnt += mk[s] ? 1 : 0;
    if (s >= 1)      acc += trans[tg[s - 1] * T_ + t] * mf;    // transition scores
    if (s <= S_ - 2) acc += lg[(size_t)s * T_ + t] * mf;       // emission scores
  }
  sred[tid] = acc; cred[tid] = cnt;
  __syncthreads();
  for (int off = 128; off > 0; off >>= 1) {
    if (tid < off) { sred[tid] += sred[tid + off]; cred[tid] += cred[tid + off]; }
    __syncthreads();
  }
  if (tid == 0) {
    int   last_idx = cred[0] - 1;
    int   lt       = tg[last_idx];
    float score    = start_t[tg[0]] + sred[0];
    score += end_t[lt] + lg[(size_t)(S_ - 1) * T_ + lt] * (mk[S_ - 1] ? 1.0f : 0.0f);
    num_out[b] = score;
  }
}

// ---------------------------------------------------------------------------
// Final: out = sum_b (num[b] - den[b]); deterministic tree reduction.
// ---------------------------------------------------------------------------
__global__ __launch_bounds__(256)
void crf_final(const float* __restrict__ num, const float* __restrict__ den,
               float* __restrict__ out)
{
  __shared__ float sred[256];
  int tid = threadIdx.x;
  sred[tid] = num[tid] - den[tid];
  __syncthreads();
  for (int off = 128; off > 0; off >>= 1) {
    if (tid < off) sred[tid] += sred[tid + off];
    __syncthreads();
  }
  if (tid == 0) out[0] = sred[0];
}

extern "C" void kernel_launch(void* const* d_in, const int* in_sizes, int n_in,
                              void* d_out, int out_size, void* d_ws, size_t ws_size,
                              hipStream_t stream) {
  const float*         inputs  = (const float*)d_in[0];
  const int*           tags    = (const int*)d_in[1];
  const unsigned char* mask    = (const unsigned char*)d_in[2];
  const float*         trans   = (const float*)d_in[3];
  const float*         start_t = (const float*)d_in[4];
  const float*         end_t   = (const float*)d_in[5];

  float* den = (float*)d_ws;        // [B_]
  float* num = den + B_;            // [B_]

  crf_forward_wmma<<<B_ / 16, 128, 0, stream>>>(inputs, mask, trans, start_t, end_t, den);
  crf_joint<<<B_, 256, 0, stream>>>(inputs, tags, mask, trans, start_t, end_t, num);
  crf_final<<<1, 256, 0, stream>>>(num, den, (float*)d_out);
}
